// IterativeBlock_15126874816567
// MI455X (gfx1250) — matrix operand order repaired and checked by reading, not compiled, not run on hardware
//
#include <hip/hip_runtime.h>
#include <math.h>

typedef __attribute__((ext_vector_type(16))) __bf16 v16bf;
typedef __attribute__((ext_vector_type(8)))  float  v8f;
typedef __attribute__((ext_vector_type(4)))  unsigned int u32x4;

#define TOKENS 16384
#define SEQ    2048
#define DMODEL 1024
#define EPSV   1.1920929e-07f

// ---------------- workspace layout (bytes) ----------------
#define OFF_H      ((size_t)0)                 // f32 [16384,1024]   64MB
#define OFF_COMB   ((size_t)67108864)          // bf16 [16384,2048]  64MB
#define OFF_G      ((size_t)134217728)         // f32 [16384,2816]  176MB (reused)
#define OFF_FFB    ((size_t)318767104)         // bf16 [16384,2816]  88MB
#define OFF_BX1    ((size_t)411041792)         // bf16 [16384,32]     1MB
#define OFF_BX2    ((size_t)412090368)         // bf16 [16384,32]     1MB
#define OFF_WPOOL  ((size_t)413138944)         // bf16 weight pool  ~4.2MB

// ---------------- WMMA operand loaders (wave32 layouts, ISA 7.12.2) -------
// A operand (16x32 bf16): lane holds row M=lane&15; K chunks [kb8,kb8+8) and
// [kb8+16,kb8+24) with kb8 = (lane>>4)*8.
__device__ __forceinline__ v16bf load_a(const __bf16* row, int lane) {
  int kb8 = (lane >> 4) << 3;
  union { struct { u32x4 lo, hi; } u; v16bf v; } c;
  c.u.lo = *(const u32x4*)(row + kb8);
  c.u.hi = *(const u32x4*)(row + kb8 + 16);
  return c.v;
}
// B operand (32x16 bf16): lane holds col N=lane&15; K chunks [ks,ks+8) and
// [ks+16,ks+24) with ks = (lane>>4)*8.  "row" points at W[o][.] (contiguous over K).
__device__ __forceinline__ v16bf load_b(const __bf16* row, int lane) {
  int ks = (lane >> 4) << 3;
  union { struct { u32x4 lo, hi; } u; v16bf v; } c;
  c.u.lo = *(const u32x4*)(row + ks);
  c.u.hi = *(const u32x4*)(row + ks + 16);
  return c.v;
}

__device__ __forceinline__ v8f wmma_bf16(v16bf a, v16bf b, v8f c) {
  return __builtin_amdgcn_wmma_f32_16x16x32_bf16(false, a, false, b,
                                                 (short)0, c, false, false);
}

__device__ __forceinline__ float sigm(float x) { return 1.0f / (1.0f + __expf(-x)); }

// ---------------- f32 -> bf16 weight conversion ----------------
__global__ void f32_to_bf16_kernel(const float* __restrict__ in,
                                   __bf16* __restrict__ out, int n) {
  int i = blockIdx.x * blockDim.x + threadIdx.x;
  if (i < n) out[i] = (__bf16)in[i];
}

// ---------------- rmsnorm (+ causal shift write) ----------------
// One block per token row. Writes z into comb[t][0:1024]; if withShift, also
// writes z into comb[t+1][1024:2048] (same sequence) and zeros comb[t][1024:]
// when t is the first token of its sequence.
__global__ void rmsnorm_kernel(const float* __restrict__ h,
                               __bf16* __restrict__ comb, int withShift) {
  __shared__ float red[256];
  int t = blockIdx.x;
  int tid = threadIdx.x;
  const float* hr = h + (size_t)t * DMODEL;
  float ss = 0.f;
  for (int i = tid; i < DMODEL; i += 256) { float v = hr[i]; ss += v * v; }
  red[tid] = ss;
  __syncthreads();
  for (int s = 128; s > 0; s >>= 1) {
    if (tid < s) red[tid] += red[tid + s];
    __syncthreads();
  }
  float scale = rsqrtf(red[0] * (1.0f / (float)DMODEL) + EPSV);
  int s_in_seq = t & (SEQ - 1);
  for (int i = tid; i < DMODEL; i += 256) {
    float z = hr[i] * scale;
    __bf16 zb = (__bf16)z;
    comb[(size_t)t * 2048 + i] = zb;
    if (withShift) {
      if (s_in_seq < SEQ - 1) comb[(size_t)(t + 1) * 2048 + 1024 + i] = zb;
      if (s_in_seq == 0)      comb[(size_t)t * 2048 + 1024 + i] = (__bf16)0.0f;
    }
  }
}

// ---------------- Bx = X @ Bw^T  (N=32, K=fin) ----------------
// One wave computes a 16M x 32N tile; output bf16 [M,32].
__global__ void bx_kernel(const __bf16* __restrict__ X, int sx, int fin,
                          const __bf16* __restrict__ Bw,
                          __bf16* __restrict__ out) {
  int lane = threadIdx.x & 31;
  int wave = threadIdx.x >> 5;
  int m_base = (blockIdx.x * 4 + wave) * 16;
  int r = lane & 15;
  v8f acc0 = {}, acc1 = {};
  const __bf16* brow0 = Bw + (size_t)r * fin;
  const __bf16* brow1 = Bw + (size_t)(16 + r) * fin;
  const __bf16* xrow  = X + (size_t)(m_base + r) * sx;
  for (int kb = 0; kb < fin; kb += 32) {
    v16bf aop = load_a(xrow + kb, lane);
    acc0 = wmma_bf16(aop, load_b(brow0 + kb, lane), acc0);
    acc1 = wmma_bf16(aop, load_b(brow1 + kb, lane), acc1);
  }
  int mo = (lane >> 4) << 3;
  int n0 = lane & 15;
#pragma unroll
  for (int rr = 0; rr < 8; ++rr) {
    int t = m_base + mo + rr;
    out[(size_t)t * 32 + n0]      = (__bf16)acc0[rr];
    out[(size_t)t * 32 + 16 + n0] = (__bf16)acc1[rr];
  }
}

// ---------------- lslinear: blockdiag(W)@x + A@(B@x), fused epilogue -------
// y[t][o] = sum_k X[t][nb*block_in+k] * W[o][k]  +  sum_j Bx[t][j] * Aw[o][j]
// wave tile: 64M x 16N (4 accumulators share the weight B-operand).
// mode: 0 outf=sigmoid(y)   1 outf += gin*y (h update)   2 outf=silu(y)
//       3 outb = bf16(gin*y)                4 outf = gin + y (final residual)
__global__ void lslin_kernel(const __bf16* __restrict__ X, int sx,
                             const __bf16* __restrict__ W,
                             const __bf16* __restrict__ Aw,
                             const __bf16* __restrict__ Bx,
                             int outdim, int block_in, int block_out,
                             int mode,
                             float* __restrict__ outf,
                             __bf16* __restrict__ outb,
                             const float* __restrict__ gin) {
  int lane = threadIdx.x & 31;
  int wave = threadIdx.x >> 5;
  int n_base = (blockIdx.x * 4 + wave) * 16;
  int m_base = blockIdx.y * 64;
  int r = lane & 15;
  int nb = n_base / block_out;          // 16 | block_out -> uniform per tile
  int xcol0 = nb * block_in;

  const __bf16* wrow = W + (size_t)(n_base + r) * block_in;
  v8f acc[4] = {};
  for (int kb = 0; kb < block_in; kb += 32) {
    v16bf bop = load_b(wrow + kb, lane);
#pragma unroll
    for (int m = 0; m < 4; ++m) {
      const __bf16* xrow =
          X + (size_t)(m_base + 16 * m + r) * sx + xcol0 + kb;
      acc[m] = wmma_bf16(load_a(xrow, lane), bop, acc[m]);
    }
  }
  { // fused rank-32 correction: one K=32 WMMA step
    const __bf16* arow = Aw + (size_t)(n_base + r) * 32;
    v16bf bop = load_b(arow, lane);
#pragma unroll
    for (int m = 0; m < 4; ++m) {
      const __bf16* bxrow = Bx + (size_t)(m_base + 16 * m + r) * 32;
      acc[m] = wmma_bf16(load_a(bxrow, lane), bop, acc[m]);
    }
  }
  int oo = n_base + r;
  int mo = (lane >> 4) << 3;
#pragma unroll
  for (int m = 0; m < 4; ++m) {
#pragma unroll
    for (int rr = 0; rr < 8; ++rr) {
      int t = m_base + 16 * m + mo + rr;
      size_t idx = (size_t)t * outdim + oo;
      float v = acc[m][rr];
      if (mode == 0)      outf[idx] = sigm(v);
      else if (mode == 1) outf[idx] += gin[idx] * v;
      else if (mode == 2) outf[idx] = v * sigm(v);
      else if (mode == 3) outb[idx] = (__bf16)(gin[idx] * v);
      else                outf[idx] = gin[idx] + v;
    }
  }
}

// ---------------- host side ----------------
extern "C" void kernel_launch(void* const* d_in, const int* in_sizes, int n_in,
                              void* d_out, int out_size, void* d_ws,
                              size_t ws_size, hipStream_t stream) {
  char* ws = (char*)d_ws;
  float*  h    = (float*)(ws + OFF_H);
  __bf16* comb = (__bf16*)(ws + OFF_COMB);
  float*  G    = (float*)(ws + OFF_G);
  __bf16* ffb  = (__bf16*)(ws + OFF_FFB);
  __bf16* bx1  = (__bf16*)(ws + OFF_BX1);
  __bf16* bx2  = (__bf16*)(ws + OFF_BX2);
  __bf16* wp   = (__bf16*)(ws + OFF_WPOOL);

  // bf16 weight pool: {input index, element offset, count}
  static const struct { int src; size_t off; int n; } cv[15] = {
      {1,       0, 262144},  // fg_W  [1024,256]
      {2,  262144,  32768},  // fg_A  [1024,32]
      {3,  294912,  65536},  // fg_B  [32,2048]
      {4,  360448, 262144},  // gu_W
      {5,  622592,  32768},  // gu_A
      {6,  655360,  65536},  // gu_B
      {7,  720896, 360448},  // ffg_W [2816,128]
      {8, 1081344,  90112},  // ffg_A [2816,32]
      {9, 1171456,  32768},  // ffg_B [32,1024]
      {10,1204224, 360448},  // fff_W
      {11,1564672,  90112},  // fff_A
      {12,1654784,  32768},  // fff_B
      {13,1687552, 360448},  // ffp_W [1024,352]
      {14,2048000,  32768},  // ffp_A [1024,32]
      {15,2080768,  90112},  // ffp_B [32,2816]
  };
  for (int i = 0; i < 15; ++i)
    f32_to_bf16_kernel<<<(cv[i].n + 255) / 256, 256, 0, stream>>>(
        (const float*)d_in[cv[i].src], wp + cv[i].off, cv[i].n);

  __bf16 *fgW = wp + cv[0].off, *fgA = wp + cv[1].off, *fgB = wp + cv[2].off;
  __bf16 *guW = wp + cv[3].off, *guA = wp + cv[4].off, *guB = wp + cv[5].off;
  __bf16 *ffgW = wp + cv[6].off, *ffgA = wp + cv[7].off, *ffgB = wp + cv[8].off;
  __bf16 *fffW = wp + cv[9].off, *fffA = wp + cv[10].off, *fffB = wp + cv[11].off;
  __bf16 *ffpW = wp + cv[12].off, *ffpA = wp + cv[13].off, *ffpB = wp + cv[14].off;

  // h = x
  hipMemcpyAsync(h, d_in[0], (size_t)TOKENS * DMODEL * sizeof(float),
                 hipMemcpyDeviceToDevice, stream);

  dim3 blk(128);                      // 4 waves / block
  dim3 gBx(TOKENS / 64);              // bx: 1 wave per 16 tokens
  dim3 g1024(1024 / 64, TOKENS / 64); // lslin outdim 1024
  dim3 g2816(2816 / 64, TOKENS / 64); // lslin outdim 2816

  for (int it = 0; it < 4; ++it) {
    rmsnorm_kernel<<<TOKENS, 256, 0, stream>>>(h, comb, 1);
    bx_kernel<<<gBx, blk, 0, stream>>>(comb, 2048, 2048, fgB, bx1);
    bx_kernel<<<gBx, blk, 0, stream>>>(comb, 2048, 2048, guB, bx2);
    // gate = sigmoid(lslinear(combined, fg)) -> G
    lslin_kernel<<<g1024, blk, 0, stream>>>(comb, 2048, fgW, fgA, bx1,
                                            1024, 256, 128, 0, G, nullptr, nullptr);
    // h += G * lslinear(combined, gu)
    lslin_kernel<<<g1024, blk, 0, stream>>>(comb, 2048, guW, guA, bx2,
                                            1024, 256, 128, 1, h, nullptr, G);
  }

  // final FF block
  rmsnorm_kernel<<<TOKENS, 256, 0, stream>>>(h, comb, 0);
  bx_kernel<<<gBx, blk, 0, stream>>>(comb, 2048, 1024, ffgB, bx1);
  bx_kernel<<<gBx, blk, 0, stream>>>(comb, 2048, 1024, fffB, bx2);
  // G = silu(lslinear(z, ffg))
  lslin_kernel<<<g2816, blk, 0, stream>>>(comb, 2048, ffgW, ffgA, bx1,
                                          2816, 128, 352, 2, G, nullptr, nullptr);
  // ff = bf16(G * lslinear(z, fff))
  lslin_kernel<<<g2816, blk, 0, stream>>>(comb, 2048, fffW, fffA, bx2,
                                          2816, 128, 352, 3, nullptr, ffb, G);
  bx_kernel<<<gBx, blk, 0, stream>>>(ffb, 2816, 2816, ffpB, bx1);
  // out = h + lslinear(ff, ffp)
  lslin_kernel<<<g1024, blk, 0, stream>>>(ffb, 2816, ffpW, ffpA, bx1,
                                          1024, 352, 128, 4, (float*)d_out,
                                          nullptr, h);
}
